// SurrogateClassifier_27625229648167
// MI455X (gfx1250) — matrix operand order repaired and checked by reading, not compiled, not run on hardware
//
#include <hip/hip_runtime.h>
#include <hip/hip_bf16.h>
#include <math.h>

// ---------------- types for WMMA (gfx1250, wave32) ----------------
typedef __bf16 bf16;
typedef bf16  bf16x8 __attribute__((ext_vector_type(8)));
typedef bf16  v16bf  __attribute__((ext_vector_type(16)));
typedef float v8f    __attribute__((ext_vector_type(8)));
typedef unsigned int u32x4 __attribute__((ext_vector_type(4)));
typedef int i32x4 __attribute__((ext_vector_type(4)));
typedef int i32x8 __attribute__((ext_vector_type(8)));

union FragU { v16bf v; bf16x8 h[2]; };

#define SLAB_K 64    // K elems staged per barrier (two 32-wide WMMA steps)
#define LROW   80    // padded LDS row stride in elems: 64 data + 2x8 TDM pad (160B)

// ---------------- Tensor Data Mover: 2D tile (64 elems x 128 rows, bf16) -> LDS ----------------
#if __has_builtin(__builtin_amdgcn_tensor_load_to_lds)
#define HAVE_TDM 1
__device__ __forceinline__ void tdm_load_tile(const bf16* gsrc, unsigned stride_elems,
                                              unsigned lds_byte_off) {
    unsigned long long ga = (unsigned long long)(uintptr_t)gsrc;
    // D# group 0: count=1 | lds_addr | global_addr[56:0] | type=2
    u32x4 g0 = { 1u,
                 lds_byte_off,
                 (unsigned)ga,
                 (unsigned)((ga >> 32) & 0x01FFFFFFu) | (2u << 30) };
    // D# group 1:
    //  dw0: wg_mask=0 | data_size=1(2B)<<16 | pad_enable<<20 | pad_interval=3(16dw=64B)<<22
    //       | pad_amount=3(4dw=16B)<<25  -> 128B rows stored at 160B stride
    //  tensor_dim0/1 = 2^30 (tiles never clip), tile = 64 x 128, dim0 stride = ld
    i32x8 g1 = { (int)((1u << 16) | (1u << 20) | (3u << 22) | (3u << 25)),
                 0,                               // abar=0 | tensor_dim0[15:0]=0
                 0x4000,                          // tensor_dim0[31:16] | tensor_dim1[15:0]=0
                 (int)(0x4000u | (64u << 16)),    // tensor_dim1[31:16] | tile_dim0=64
                 128,                             // tile_dim1=128 | tile_dim2=0
                 (int)stride_elems,               // tensor_dim0_stride[31:0]
                 0, 0 };
    i32x4 gz = { 0, 0, 0, 0 };
#if defined(__clang_major__) && (__clang_major__ >= 23)
    i32x8 gz8 = { 0, 0, 0, 0, 0, 0, 0, 0 };
    __builtin_amdgcn_tensor_load_to_lds(g0, g1, gz, gz, gz8, 0);
#else
    __builtin_amdgcn_tensor_load_to_lds(g0, g1, gz, gz, 0);
#endif
}
#else
#define HAVE_TDM 0
#endif

// fallback cooperative copy (same padded layout), 128 threads
__device__ __forceinline__ void coop_copy_tile(const bf16* __restrict__ gsrc, int ld,
                                               bf16* __restrict__ dst) {
    for (int c = threadIdx.x; c < 1024; c += 128) {      // 128 rows x 8 x 16B
        int row = c >> 3, seg = c & 7;
        *(bf16x8*)(dst + row * LROW + (seg >> 2) * 40 + (seg & 3) * 8) =
            *(const bf16x8*)(gsrc + (size_t)row * ld + seg * 8);
    }
}

// A fragment 16x32 (MxK): lane<16 -> row, K +0..7 / +16..23 ; lane>=16 -> +8 / +24
__device__ __forceinline__ v16bf lds_a_frag(const bf16* t, int row, int ko, bool lo) {
    FragU f;
    const bf16* p = t + row * LROW + ko + (lo ? 0 : 8);
    f.h[0] = *(const bf16x8*)(p);
    f.h[1] = *(const bf16x8*)(p + 16);
    return f.v;
}
// B fragment 32x16 (KxN) from [N,K]-major tile: lane<16 -> col, K +0..7 / +16..23 ; lane>=16 -> +8 / +24
__device__ __forceinline__ v16bf lds_b_frag(const bf16* t, int row, int ko, bool lo) {
    FragU f;
    const bf16* p = t + row * LROW + ko + (lo ? 0 : 8);
    f.h[0] = *(const bf16x8*)(p);
    f.h[1] = *(const bf16x8*)(p + 16);
    return f.v;
}

// ---------------- WMMA GEMM: C[M,N] = A[M,K] @ B[N,K]^T ----------------
// MODE 0: Cf[row,col] = acc + (bias && col<nvalid ? bias[col] : 0)
// MODE 1: featout[row,col] = bf16( g[row]*xin[row,col] + (1-g[row])*acc )
// Block = 128 threads = 4 waves (2x2), block tile 128x128, wave tile 64x64.
// K staged in LDS as 64-wide slabs via double-buffered TDM loads issued by wave 0.
template <int MODE>
__global__ __launch_bounds__(128) void gemm_wmma_bf16(
    const bf16* __restrict__ A, const bf16* __restrict__ B,
    int K, int lda, int ldb,
    float* __restrict__ Cf, int ldc, const float* __restrict__ bias, int nvalid,
    const float* __restrict__ g, const float* __restrict__ xin, bf16* __restrict__ featout)
{
    __shared__ bf16 ldsA[2][128 * LROW];
    __shared__ bf16 ldsB[2][128 * LROW];

    const int lane = threadIdx.x & 31;
    const int wave = threadIdx.x >> 5;
    const int wr = wave >> 1;                 // 0..1 -> M within block tile
    const int wc = wave & 1;                  // 0..1 -> N within block tile
    const int bm = blockIdx.y * 128, bn = blockIdx.x * 128;
    const int m0 = wr * 64, n0 = wc * 64;     // origins inside the LDS tile
    const bool lo = lane < 16;
    const int  l15 = lane & 15;

    const bf16* Ablk = A + (size_t)bm * lda;
    const bf16* Bblk = B + (size_t)bn * ldb;

    v8f acc[4][4];
#pragma unroll
    for (int i = 0; i < 4; ++i)
#pragma unroll
        for (int j = 0; j < 4; ++j) acc[i][j] = (v8f){};

#if HAVE_TDM
    if (wave == 0) {
        tdm_load_tile(Ablk, (unsigned)lda, (unsigned)(uintptr_t)&ldsA[0][0]);
        tdm_load_tile(Bblk, (unsigned)ldb, (unsigned)(uintptr_t)&ldsB[0][0]);
    }
#endif

    int cur = 0;
    for (int k0 = 0; k0 < K; k0 += SLAB_K) {
#if HAVE_TDM
        if (wave == 0) __builtin_amdgcn_s_wait_tensorcnt(0);
        __syncthreads();  // cur buffer ready for all; cur^1 fully consumed by all
        if (wave == 0 && k0 + SLAB_K < K) {
            tdm_load_tile(Ablk + k0 + SLAB_K, (unsigned)lda,
                          (unsigned)(uintptr_t)&ldsA[cur ^ 1][0]);
            tdm_load_tile(Bblk + k0 + SLAB_K, (unsigned)ldb,
                          (unsigned)(uintptr_t)&ldsB[cur ^ 1][0]);
        }
#else
        __syncthreads();
        coop_copy_tile(Ablk + k0, lda, &ldsA[cur][0]);
        coop_copy_tile(Bblk + k0, ldb, &ldsB[cur][0]);
        __syncthreads();
#endif
        const bf16* At = &ldsA[cur][0];
        const bf16* Bt = &ldsB[cur][0];
#pragma unroll
        for (int kk = 0; kk < 2; ++kk) {      // two 32-wide K steps per slab
            const int ko = kk * 40;           // second data chunk starts after 16B pad
            v16bf a[4];
#pragma unroll
            for (int i = 0; i < 4; ++i) a[i] = lds_a_frag(At, m0 + i * 16 + l15, ko, lo);
#pragma unroll
            for (int j = 0; j < 4; ++j) {
                v16bf b = lds_b_frag(Bt, n0 + j * 16 + l15, ko, lo);
#pragma unroll
                for (int i = 0; i < 4; ++i) {
                    acc[i][j] = __builtin_amdgcn_wmma_f32_16x16x32_bf16(
                        false, a[i], false, b, (short)0, acc[i][j], false, false);
                }
            }
        }
        cur ^= 1;
    }

    // C layout: VGPR r -> row tile*16 + r + (lane<16 ? 0 : 8), col = tile*16 + (lane&15)
#pragma unroll
    for (int i = 0; i < 4; ++i) {
#pragma unroll
        for (int j = 0; j < 4; ++j) {
            const int col = bn + n0 + j * 16 + l15;
#pragma unroll
            for (int r = 0; r < 8; ++r) {
                const int row = bm + m0 + i * 16 + r + (lo ? 0 : 8);
                float v = acc[i][j][r];
                if (MODE == 0) {
                    float bv = (bias && col < nvalid) ? bias[col] : 0.f;
                    Cf[(size_t)row * ldc + col] = v + bv;
                } else {
                    float gv = g[row];
                    float xv = xin[(size_t)row * ldc + col];
                    featout[(size_t)row * ldc + col] = (bf16)(gv * xv + (1.f - gv) * v);
                }
            }
        }
    }
}

// ---------------- elementwise / reduction kernels ----------------
__global__ void f32_to_bf16_pad(const float* __restrict__ src, bf16* __restrict__ dst,
                                size_t total, size_t valid) {
    size_t i = (size_t)blockIdx.x * blockDim.x + threadIdx.x;
    if (i < total) dst[i] = (i < valid) ? (bf16)src[i] : (bf16)0.f;
}

__global__ void segsum_kernel(const float* __restrict__ x, const int* __restrict__ labels,
                              float* __restrict__ sums, float* __restrict__ cnts, int D) {
    int n = blockIdx.x;
    int lbl = labels[n];
    if (lbl < 0) return;
    if (threadIdx.x == 0) atomicAdd(&cnts[lbl], 1.0f);
    const float* xr = x + (size_t)n * D;
    float* sr = sums + (size_t)lbl * D;
    for (int d = threadIdx.x; d < D; d += blockDim.x) atomicAdd(&sr[d], xr[d]);
}

// v^T[d, c] with bias-corrected EMA; c == C-1 is the zero pad column.
__global__ void compute_vT(const float* __restrict__ sums, const float* __restrict__ cnts,
                           const float* __restrict__ hist, const int* __restrict__ hist_cnt,
                           bf16* __restrict__ vT, int C, int D) {
    int c = blockIdx.x;
    if (c == C - 1) {
        for (int d = threadIdx.x; d < D; d += blockDim.x) vT[(size_t)d * C + c] = (bf16)0.f;
        return;
    }
    float cnt = cnts[c + 1];
    bool present = cnt > 0.f;
    int ncount = hist_cnt[c] + (present ? 1 : 0);
    int clamped = ncount < 1 ? 1 : ncount;
    float scale = 0.1f / (1.f - powf(0.9f, (float)clamped));
    float inv = 1.f / fmaxf(cnt, 1.f);
    for (int d = threadIdx.x; d < D; d += blockDim.x) {
        float mean = sums[(size_t)(c + 1) * D + d] * inv;
        float h = hist[(size_t)c * D + d];
        float nh = present ? 0.9f * h + mean : h;
        vT[(size_t)d * C + c] = (bf16)(nh * scale);
    }
}

__global__ void compute_g(const float* __restrict__ x, const float* __restrict__ wg,
                          const float* __restrict__ bg, float* __restrict__ g, int D) {
    __shared__ float red[256];
    int n = blockIdx.x;
    float s = 0.f;
    for (int d = threadIdx.x; d < D; d += blockDim.x) s += x[(size_t)n * D + d] * wg[d];
    red[threadIdx.x] = s; __syncthreads();
    for (int t = 128; t > 0; t >>= 1) {
        if (threadIdx.x < t) red[threadIdx.x] += red[threadIdx.x + t];
        __syncthreads();
    }
    if (threadIdx.x == 0) g[n] = 1.f / (1.f + expf(-(red[0] + bg[0])));
}

__global__ void softmax_to_bf16(const float* __restrict__ logits, bf16* __restrict__ wbf,
                                int ld, int nvalid) {
    __shared__ float red[256];
    int n = blockIdx.x;
    const float* row = logits + (size_t)n * ld;
    float mx = -__builtin_inff();
    for (int c = threadIdx.x; c < nvalid; c += blockDim.x) mx = fmaxf(mx, row[c]);
    red[threadIdx.x] = mx; __syncthreads();
    for (int t = 128; t > 0; t >>= 1) {
        if (threadIdx.x < t) red[threadIdx.x] = fmaxf(red[threadIdx.x], red[threadIdx.x + t]);
        __syncthreads();
    }
    mx = red[0]; __syncthreads();
    float s = 0.f;
    for (int c = threadIdx.x; c < nvalid; c += blockDim.x) s += expf(row[c] - mx);
    red[threadIdx.x] = s; __syncthreads();
    for (int t = 128; t > 0; t >>= 1) {
        if (threadIdx.x < t) red[threadIdx.x] += red[threadIdx.x + t];
        __syncthreads();
    }
    float invs = 1.f / red[0];
    bf16* wrow = wbf + (size_t)n * ld;
    for (int c = threadIdx.x; c < ld; c += blockDim.x)
        wrow[c] = (c < nvalid) ? (bf16)(expf(row[c] - mx) * invs) : (bf16)0.f;
}

__global__ void ce_rows(const float* __restrict__ logits, const int* __restrict__ labels,
                        float* __restrict__ lacc, int C) {
    __shared__ float red[256];
    int n = blockIdx.x;
    const float* row = logits + (size_t)n * C;
    float mx = -__builtin_inff();
    for (int c = threadIdx.x; c < C; c += blockDim.x) mx = fmaxf(mx, row[c]);
    red[threadIdx.x] = mx; __syncthreads();
    for (int t = 128; t > 0; t >>= 1) {
        if (threadIdx.x < t) red[threadIdx.x] = fmaxf(red[threadIdx.x], red[threadIdx.x + t]);
        __syncthreads();
    }
    mx = red[0]; __syncthreads();
    float s = 0.f;
    for (int c = threadIdx.x; c < C; c += blockDim.x) s += expf(row[c] - mx);
    red[threadIdx.x] = s; __syncthreads();
    for (int t = 128; t > 0; t >>= 1) {
        if (threadIdx.x < t) red[threadIdx.x] += red[threadIdx.x + t];
        __syncthreads();
    }
    if (threadIdx.x == 0) {
        int lbl = labels[n];
        if (lbl >= 0) {
            float nll = -(row[lbl] - mx - logf(red[0]));
            atomicAdd(&lacc[0], nll);
            atomicAdd(&lacc[1], 1.0f);
        }
    }
}

__global__ void finalize_loss(const float* __restrict__ lacc, float* __restrict__ out) {
    out[0] = lacc[0] / fmaxf(lacc[1], 1.f);
}

// ---------------- host launch ----------------
extern "C" void kernel_launch(void* const* d_in, const int* in_sizes, int n_in,
                              void* d_out, int out_size, void* d_ws, size_t ws_size,
                              hipStream_t stream) {
    constexpr int N = 8192, D = 1024, C = 8192;
    constexpr int CV = C - 1;

    const float* x      = (const float*)d_in[0];
    const float* x_ctx  = (const float*)d_in[1];
    const int*   labels = (const int*)d_in[2];
    const float* cls    = (const float*)d_in[3];
    const float* qw     = (const float*)d_in[4];
    const float* qb     = (const float*)d_in[5];
    const float* wg     = (const float*)d_in[6];
    const float* bg     = (const float*)d_in[7];
    const float* hist   = (const float*)d_in[8];
    const int*   hcnt   = (const int*)d_in[9];

    float* out   = (float*)d_out;       // out[0] = loss, out[1..] = logits [N, C]
    float* wlog  = out + 1;             // reuse logits region as f32 w_logits scratch

    char* ws = (char*)d_ws;
    constexpr size_t SUMS_B = (size_t)C * D * 4;
    constexpr size_t OFF_SUMS = 0;
    constexpr size_t OFF_CNTS = OFF_SUMS + SUMS_B;
    constexpr size_t OFF_LOSS = OFF_CNTS + (size_t)C * 4;
    constexpr size_t OFF_G    = OFF_LOSS + 256;
    constexpr size_t OFF_XCBF = OFF_G + (size_t)N * 4;
    constexpr size_t BFMAT    = (size_t)N * D * 2;
    constexpr size_t OFF_QWBF = OFF_XCBF + BFMAT;
    constexpr size_t OFF_CLSBF= OFF_QWBF + BFMAT;
    constexpr size_t OFF_VTBF = OFF_CLSBF + BFMAT;
    constexpr size_t OFF_FEAT = OFF_VTBF + (size_t)D * C * 2;
    constexpr size_t OFF_WBF  = OFF_FEAT + BFMAT;

    float* sums  = (float*)(ws + OFF_SUMS);
    float* cnts  = (float*)(ws + OFF_CNTS);
    float* lacc  = (float*)(ws + OFF_LOSS);
    float* gbuf  = (float*)(ws + OFF_G);
    bf16*  xcbf  = (bf16*)(ws + OFF_XCBF);
    bf16*  qwbf  = (bf16*)(ws + OFF_QWBF);
    bf16*  clsbf = (bf16*)(ws + OFF_CLSBF);
    bf16*  vtbf  = (bf16*)(ws + OFF_VTBF);
    bf16*  featbf= (bf16*)(ws + OFF_FEAT);
    bf16*  wbf   = (bf16*)(ws + OFF_WBF);

    hipMemsetAsync(ws, 0, OFF_LOSS + 256, stream);

    {
        const size_t total = (size_t)N * D;
        dim3 g((unsigned)((total + 255) / 256));
        f32_to_bf16_pad<<<g, 256, 0, stream>>>(x_ctx, xcbf,  total, total);
        f32_to_bf16_pad<<<g, 256, 0, stream>>>(cls,   clsbf, total, total);
        f32_to_bf16_pad<<<g, 256, 0, stream>>>(qw,    qwbf,  total, (size_t)CV * D);
    }

    segsum_kernel<<<N, 256, 0, stream>>>(x, labels, sums, cnts, D);
    compute_vT<<<C, 256, 0, stream>>>(sums, cnts, hist, hcnt, vtbf, C, D);
    compute_g<<<N, 256, 0, stream>>>(x, wg, bg, gbuf, D);

    // GEMM1: w_logits[N, C] = x_ctx @ qw^T + qb   (pad col -> 0)
    {
        dim3 grid(C / 128, N / 128);
        gemm_wmma_bf16<0><<<grid, 128, 0, stream>>>(
            xcbf, qwbf, D, D, D, wlog, C, qb, CV, nullptr, nullptr, nullptr);
    }
    softmax_to_bf16<<<N, 256, 0, stream>>>(wlog, wbf, C, CV);

    // GEMM2: substitute = w @ v ; fused epilogue feat = g*x + (1-g)*sub -> bf16
    {
        dim3 grid(D / 128, N / 128);
        gemm_wmma_bf16<1><<<grid, 128, 0, stream>>>(
            wbf, vtbf, C, C, C, nullptr, D, nullptr, 0, gbuf, x, featbf);
    }

    // GEMM3: logits[N, C] = feat @ classifier^T  (overwrites w_logits scratch)
    {
        dim3 grid(C / 128, N / 128);
        gemm_wmma_bf16<0><<<grid, 128, 0, stream>>>(
            featbf, clsbf, D, D, D, wlog, C, nullptr, 0, nullptr, nullptr, nullptr);
    }

    ce_rows<<<N, 256, 0, stream>>>(wlog, labels, lacc, C);
    finalize_loss<<<1, 1, 0, stream>>>(lacc, out);
}
